// SpiralGrid_48756468744572
// MI455X (gfx1250) — matrix operand order repaired and checked by reading, compiled, not run on hardware
//
#include <hip/hip_runtime.h>
#include <hip/hip_bf16.h>
#include <cstdint>

typedef __attribute__((ext_vector_type(16))) _Float16 v16h;
typedef __attribute__((ext_vector_type(8)))  _Float16 v8h;
typedef __attribute__((ext_vector_type(8)))  float    v8f;
typedef __attribute__((ext_vector_type(4)))  float    f32x4;

#define GH   32
#define GW   32
#define NCELL 1024          // H*W
#define CCH  256            // C
#define HID_ 512
#define TWOC 512            // 2*C
#define BATCH 128
#define BSUB 16             // batch rows per workgroup (one WMMA M-tile)
#define NWG  (BATCH / BSUB) // 8 workgroups
#define NTHREADS 512        // 16 waves, 4 waves/SIMD -> ~256 VGPR budget
#define KT_REG 8            // K-tiles 0..7   : register-resident B
#define KT_LDS 8            // K-tiles 8..15  : LDS-resident B (256 KB)

// LDS partition (dynamic): [0,16K) u_f16[16][512]; [16K,16K+256K) B tiles; then y[16]
#define U_BYTES   (BSUB * TWOC * 2)               // 16384
#define BLDS_BYTES (32 * KT_LDS * 512 * 2)        // 262144
#define Y_OFF     (U_BYTES + BLDS_BYTES)          // 278528
#define SMEM_BYTES (Y_OFF + 64)

// ---------------- prep: w1 (2C x HID) f32 row-major -> w1t (HID x 2C) f16 ----------------
__global__ void prep_w1t(const float* __restrict__ w1, _Float16* __restrict__ w1t) {
  int idx = blockIdx.x * blockDim.x + threadIdx.x;   // 0 .. 262143
  int k = idx & (TWOC - 1);
  int n = idx >> 9;
  w1t[(size_t)n * TWOC + k] = (_Float16)w1[(size_t)k * HID_ + n];
}

// ---------------- prep: spiral cell order (matches reference spiral_indices) ----------------
__global__ void prep_spiral(int* __restrict__ pidx) {
  if (threadIdx.x | blockIdx.x) return;
  bool seen[NCELL];
  for (int i = 0; i < NCELL; ++i) seen[i] = false;
  int n = 0;
  auto add = [&](int y, int x) {
    if (y >= 0 && y < GH && x >= 0 && x < GW) {
      int p = y * GW + x;
      if (!seen[p]) { seen[p] = true; pidx[n++] = p; }
    }
  };
  const int cy = GH / 2, cx = GW / 2;
  add(cy, cx);
  for (int r = 1; r <= 16; ++r) {
    int top = cy - r, bot = cy + r, left = cx - r, right = cx + r;
    for (int x = left; x <= right; ++x) add(top, x);
    for (int y = top + 1; y <= bot; ++y) add(y, right);
    for (int x = right - 1; x >= left; --x) add(bot, x);
    for (int y = bot - 1; y >= top + 1; --y) add(y, left);
  }
}

// ---------------- prep: g = x (working state lives in d_out, L2-resident) ----------------
__global__ void copy_x(const f32x4* __restrict__ x, f32x4* __restrict__ g, int n4) {
  int i = blockIdx.x * blockDim.x + threadIdx.x;
  int stride = gridDim.x * blockDim.x;
  for (; i < n4; i += stride) g[i] = x[i];
}

// ---------------- main: per-workgroup sequential spiral scan over its 16 batch rows ----------------
__global__ __launch_bounds__(NTHREADS) void spiral_scan(
    float* __restrict__ g, const _Float16* __restrict__ w1t,
    const int* __restrict__ pidx,
    const float* __restrict__ b1, const float* __restrict__ w2,
    const float* __restrict__ b2, const float* __restrict__ w_y) {
  extern __shared__ __align__(32) char smem[];
  _Float16* u_lds = (_Float16*)smem;                 // 16 x 512 f16
  _Float16* b_lds = (_Float16*)(smem + U_BYTES);     // [nt(32)][kt8(8)][512 halves]
  float*    ylds  = (float*)(smem + Y_OFF);          // [16]

  const int tid  = threadIdx.x;
  const int lane = tid & 31;
  const int wave = tid >> 5;          // 0..15, each owns 32 HID columns (2 N-tiles)
  const int hi   = lane >> 4;
  const int lm   = lane & 15;
  const int b0 = blockIdx.x * BSUB;
  const int n0 = wave * 32 + lm;      // slab 0 column
  const int n1 = n0 + 16;             // slab 1 column

  // ---- one-time: B operand staging ----
  // B layout (32x16 f16 tile): lane l -> n = l%16; element e -> k = (l/16)*16 + e.
  v16h Breg0[KT_REG], Breg1[KT_REG];  // K-tiles 0..7, both slabs: 128 VGPRs
  {
    const _Float16* wp0 = w1t + (size_t)n0 * TWOC + hi * 16;
    const _Float16* wp1 = w1t + (size_t)n1 * TWOC + hi * 16;
#pragma unroll
    for (int kt = 0; kt < KT_REG; ++kt) {
      Breg0[kt] = *(const v16h*)(wp0 + kt * 32);
      Breg1[kt] = *(const v16h*)(wp1 + kt * 32);
    }
  }
  // K-tiles 8..15 for all 32 slabs -> LDS, per-lane-contiguous (tile*512 + l*16 halves)
  if (tid < 256) {
    const int nt  = tid >> 3;         // 0..31
    const int kt8 = tid & 7;          // 0..7  (real kt = kt8 + 8)
    _Float16* dst = b_lds + ((size_t)(nt * KT_LDS + kt8) << 9);
#pragma unroll 4
    for (int l = 0; l < 32; ++l) {
      const _Float16* src = w1t + (size_t)(nt * 16 + (l & 15)) * TWOC
                          + (kt8 + 8) * 32 + (l >> 4) * 16;
      *(v16h*)(dst + l * 16) = *(const v16h*)src;
    }
  }
  const float b1v0 = b1[n0], b1v1 = b1[n1];
  const float w2v0 = w2[n0], w2v1 = w2[n1];
  const float b2v  = b2[0];

  // phase A/C geometry: 32 threads per batch row, 8 channels each
  const int m  = tid >> 5;            // batch row 0..15
  const int ch = (tid & 31) << 3;     // channel 0..248 step 8
  float* rowbase = g + ((size_t)(b0 + m)) * NCELL * CCH;

  __syncthreads();                    // B-LDS ready

  for (int i = 0; i < NCELL; ++i) {
    const int cell = pidx[i];
    const int cy = cell >> 5, cx = cell & 31;

    // ---- phase A: u = [local, mean(neighbors)] -> f16 in LDS ----
    if (tid < BSUB) ylds[tid] = b2v;
    {
      const float* cbase = rowbase + (size_t)cell * CCH + ch;
      f32x4 la = *(const f32x4*)cbase;
      f32x4 lb = *(const f32x4*)(cbase + 4);
      f32x4 sa = {0.f, 0.f, 0.f, 0.f}, sb = {0.f, 0.f, 0.f, 0.f};
      float cnt = 0.f;
      if (cy > 0)      { sa += *(const f32x4*)(cbase - GW * CCH);
                         sb += *(const f32x4*)(cbase - GW * CCH + 4); cnt += 1.f; }
      if (cy < GH - 1) { sa += *(const f32x4*)(cbase + GW * CCH);
                         sb += *(const f32x4*)(cbase + GW * CCH + 4); cnt += 1.f; }
      if (cx > 0)      { sa += *(const f32x4*)(cbase - CCH);
                         sb += *(const f32x4*)(cbase - CCH + 4); cnt += 1.f; }
      if (cx < GW - 1) { sa += *(const f32x4*)(cbase + CCH);
                         sb += *(const f32x4*)(cbase + CCH + 4); cnt += 1.f; }
      const float inv = 1.0f / fmaxf(cnt, 1.0f);
      v8h lv, nv;
#pragma unroll
      for (int q = 0; q < 4; ++q) {
        lv[q]     = (_Float16)la[q];
        lv[q + 4] = (_Float16)lb[q];
        nv[q]     = (_Float16)(sa[q] * inv);
        nv[q + 4] = (_Float16)(sb[q] * inv);
      }
      *(v8h*)(&u_lds[m * TWOC + ch])       = lv;   // local half
      *(v8h*)(&u_lds[m * TWOC + CCH + ch]) = nv;   // neighbor half
    }
    __syncthreads();

    // ---- phase B: (16x512)@(512x512) via v_wmma_f32_16x16x32_f16 ----
    // A layout: lane l -> m = l%16; element e -> k = kt*32 + hi*16 + e
    v8f acc0 = {}, acc1 = {};
    {
      const _Float16* arow = u_lds + lm * TWOC + hi * 16;
#pragma unroll
      for (int kt = 0; kt < KT_REG; ++kt) {
        v8h alo = *(const v8h*)(arow + kt * 32);
        v8h ahi = *(const v8h*)(arow + kt * 32 + 8);
        v16h a = __builtin_shufflevector(alo, ahi,
            0, 1, 2, 3, 4, 5, 6, 7, 8, 9, 10, 11, 12, 13, 14, 15);
        acc0 = __builtin_amdgcn_wmma_f32_16x16x32_f16(
            false, a, false, Breg0[kt], (short)0, acc0, false, false);
        acc1 = __builtin_amdgcn_wmma_f32_16x16x32_f16(
            false, a, false, Breg1[kt], (short)0, acc1, false, false);
      }
      const _Float16* bt0 = b_lds + ((size_t)(wave * 2) * KT_LDS << 9) + lane * 16;
      const _Float16* bt1 = bt0 + ((size_t)KT_LDS << 9);
#pragma unroll
      for (int kt8 = 0; kt8 < KT_LDS; ++kt8) {
        const int kt = kt8 + 8;
        v8h alo = *(const v8h*)(arow + kt * 32);
        v8h ahi = *(const v8h*)(arow + kt * 32 + 8);
        v16h a = __builtin_shufflevector(alo, ahi,
            0, 1, 2, 3, 4, 5, 6, 7, 8, 9, 10, 11, 12, 13, 14, 15);
        v16h bv0 = *(const v16h*)(bt0 + (kt8 << 9));
        v16h bv1 = *(const v16h*)(bt1 + (kt8 << 9));
        acc0 = __builtin_amdgcn_wmma_f32_16x16x32_f16(
            false, a, false, bv0, (short)0, acc0, false, false);
        acc1 = __builtin_amdgcn_wmma_f32_16x16x32_f16(
            false, a, false, bv1, (short)0, acc1, false, false);
      }
    }
    // D layout: VGPR r, lane -> row mr = r + 8*hi. Fuse tanh(+b1)*w2, reduce over columns.
#pragma unroll
    for (int r = 0; r < 8; ++r) {
      float t = tanhf(acc0[r] + b1v0) * w2v0 + tanhf(acc1[r] + b1v1) * w2v1;
      t += __shfl_xor(t, 1);
      t += __shfl_xor(t, 2);
      t += __shfl_xor(t, 4);
      t += __shfl_xor(t, 8);          // sum over the 16-lane half (fixed mr)
      if (lm == 0) atomicAdd(&ylds[(hi << 3) + r], t);
    }
    __syncthreads();

    // ---- phase C: g[cell] += y * w_y, prefetch next cell ----
    {
      const float ym = ylds[m];
      float* dst = rowbase + (size_t)cell * CCH + ch;
      f32x4 wa = *(const f32x4*)(w_y + ch);
      f32x4 wb = *(const f32x4*)(w_y + ch + 4);
      f32x4 ca = *(f32x4*)dst;
      f32x4 cb = *(f32x4*)(dst + 4);
      ca += ym * wa;
      cb += ym * wb;
      *(f32x4*)dst       = ca;
      *(f32x4*)(dst + 4) = cb;
      if (i + 1 < NCELL) {
        const int ncell = pidx[i + 1];
        __builtin_prefetch(rowbase + (size_t)ncell * CCH + ch, 0, 1);
      }
    }
    __threadfence_block();   // make g updates visible WGP-wide before next step's reads
    __syncthreads();
  }
}

extern "C" void kernel_launch(void* const* d_in, const int* in_sizes, int n_in,
                              void* d_out, int out_size, void* d_ws, size_t ws_size,
                              hipStream_t stream) {
  (void)in_sizes; (void)n_in; (void)out_size; (void)ws_size;
  const float* x   = (const float*)d_in[0];
  const float* w1  = (const float*)d_in[1];
  const float* b1  = (const float*)d_in[2];
  const float* w2  = (const float*)d_in[3];
  const float* b2  = (const float*)d_in[4];
  const float* w_y = (const float*)d_in[5];
  float* g = (float*)d_out;

  _Float16* w1t = (_Float16*)d_ws;                                        // 512 KB
  int* pidx = (int*)((char*)d_ws + (size_t)HID_ * TWOC * sizeof(_Float16)); // + 4 KB

  prep_w1t   <<<dim3(512),  dim3(512), 0, stream>>>(w1, w1t);
  prep_spiral<<<dim3(1),    dim3(32),  0, stream>>>(pidx);
  const int n4 = BATCH * NCELL * CCH / 4;
  copy_x     <<<dim3(4096), dim3(256), 0, stream>>>((const f32x4*)x, (f32x4*)g, n4);
  spiral_scan<<<dim3(NWG), dim3(NTHREADS), SMEM_BYTES, stream>>>(
      g, w1t, pidx, b1, w2, b2, w_y);
}
